// TransformerEncoderLayer_48704929136863
// MI455X (gfx1250) — matrix operand order repaired and checked by reading, not compiled, not run on hardware
//
#include <hip/hip_runtime.h>
#include <hip/hip_bf16.h>
#include <math.h>

// ---- problem constants (match reference) ----
#define D_MODEL 1024
#define HEADS   16
#define D_K     64
#define D_FF    4096
#define SEQ     2048
#define BATCH   2
#define NROWS   (BATCH * SEQ)   // 4096 rows of [*, D_MODEL]

typedef __bf16 bf16_t;
typedef __attribute__((ext_vector_type(8)))  __bf16    v8bf;
typedef __attribute__((ext_vector_type(16))) __bf16    v16bf;
typedef __attribute__((ext_vector_type(8)))  float     v8f;
typedef __attribute__((ext_vector_type(4)))  unsigned  v4u;
typedef __attribute__((ext_vector_type(8)))  unsigned  v8u;

static __device__ __forceinline__ v16bf make_frag(v8bf lo, v8bf hi) {
  return __builtin_shufflevector(lo, hi, 0,1,2,3,4,5,6,7,8,9,10,11,12,13,14,15);
}

// Generic->LDS byte offset: on CDNA5 the low 32 bits of a flat LDS address are
// the LDS offset (ISA 10.2 aperture mapping).
static __device__ __forceinline__ unsigned lds_off(const void* p) {
  return (unsigned)(size_t)p;
}

// ------------------------------------------------------------
// Tensor Data Mover: 2-D tile load Global->LDS (ISA ch.8 D# layout).
// data_size = 2 bytes (bf16). tile = tile0 x tile1, row stride = stride0
// elements. Issued once per wave (EXEC ignored); tracked by TENSORcnt.
// ------------------------------------------------------------
static __device__ __forceinline__ void tdm_load_2d_bf16(
    unsigned lds_addr, unsigned long long gaddr,
    unsigned tdim0, unsigned tdim1, unsigned tile0, unsigned tile1,
    unsigned stride0) {
  v4u g0;
  g0.x = 1u;                                   // count=1, user mode, no gather
  g0.y = lds_addr;                             // LDS byte address
  g0.z = (unsigned)gaddr;                      // global_addr[31:0]
  g0.w = (unsigned)(gaddr >> 32) | (2u << 30); // global_addr[56:32] | type=2
  v8u g1;
  g1.s0 = 0x00010000u;                         // wg_mask=0, data_size=1 (2B)
  g1.s1 = (tdim0 & 0xFFFFu) << 16;             // tensor_dim0[15:0]
  g1.s2 = (tdim0 >> 16) | ((tdim1 & 0xFFFFu) << 16);
  g1.s3 = (tdim1 >> 16) | (tile0 << 16);       // tile_dim0
  g1.s4 = tile1;                               // tile_dim1 (tile_dim2 = 0)
  g1.s5 = stride0;                             // tensor_dim0_stride[31:0]
  g1.s6 = 0u;                                  // stride0[47:32] | stride1 lo
  g1.s7 = 0u;
  asm volatile("tensor_load_to_lds %0, %1" :: "s"(g0), "s"(g1) : "memory");
}

// Per-lane async 16B copy Global->LDS; IOFFSET applies to both addresses.
#define ASYNC_LDS_B128(ldsdst, gsrc, OFF)                                   \
  asm volatile("global_load_async_to_lds_b128 %0, %1, off offset:" #OFF    \
               :: "v"(ldsdst), "v"(gsrc) : "memory")
#define ASYNC_WAIT0() asm volatile("s_wait_asynccnt 0x0" ::: "memory")

// ============================================================
// prep: qkin = bf16(src + pos); srcb = bf16(src)
// ============================================================
__global__ __launch_bounds__(256)
void prep_kernel(const float* __restrict__ src, const float* __restrict__ pos,
                 bf16_t* __restrict__ qkin, bf16_t* __restrict__ srcb) {
  size_t i = (size_t)blockIdx.x * 256 + threadIdx.x;
  float s = src[i];
  qkin[i] = (bf16_t)(s + pos[i]);
  srcb[i] = (bf16_t)s;
}

// ============================================================
// transpose-convert: Wt[n*K + k] = bf16(W[k*N + n]);  W is [K, N]
// ============================================================
__global__ __launch_bounds__(256)
void transpose_kernel(const float* __restrict__ W, bf16_t* __restrict__ Wt,
                      int K, int N) {
  size_t i = (size_t)blockIdx.x * 256 + threadIdx.x;   // over N*K
  size_t n = i / (size_t)K;
  size_t k = i % (size_t)K;
  Wt[i] = (bf16_t)W[k * (size_t)N + n];
}

// ============================================================
// GEMM: C[M,N] = A[M,K](bf16 row-major) @ Wt[N,K](bf16 = W^T) + bias
// 256 threads = 8 waves; block tile 128x128; per-wave 32x64 (2x4 WMMA accs).
// K stepped by 32; tiles staged by the Tensor Data Mover, double-buffered.
// ============================================================
template<bool OUT_BF16, bool RELU>
__global__ __launch_bounds__(256)
void gemm_bf16_kernel(const bf16_t* __restrict__ A, const bf16_t* __restrict__ Wt,
                      const float* __restrict__ bias,
                      float* __restrict__ Cf, bf16_t* __restrict__ Cb,
                      int M, int N, int K) {
  __shared__ __align__(128) bf16_t lA[2][128][32];
  __shared__ __align__(128) bf16_t lB[2][128][32];
  const int tid   = threadIdx.x;
  const int lane  = tid & 31;
  const int lanen = lane & 15;
  const int half  = lane >> 4;
  const int wave  = tid >> 5;
  const int waveM = wave & 3;   // 4 waves along M (32 rows each)
  const int waveN = wave >> 2;  // 2 waves along N (64 cols each)
  const int tileM = blockIdx.y * 128;
  const int tileN = blockIdx.x * 128;

  const unsigned long long gA = (unsigned long long)(size_t)A;
  const unsigned long long gB = (unsigned long long)(size_t)Wt;

  const v8f vzero = {0.f,0.f,0.f,0.f,0.f,0.f,0.f,0.f};
  v8f acc[2][4];
  for (int mi = 0; mi < 2; ++mi)
    for (int ni = 0; ni < 4; ++ni) acc[mi][ni] = vzero;

  // stage k-chunk kb into LDS buffer `buf` via TDM (wave 0 only)
  auto stage = [&](int kb, int buf) {
    if (wave == 0) {
      tdm_load_2d_bf16(lds_off(&lA[buf][0][0]),
                       gA + ((size_t)tileM * K + kb) * 2,
                       (unsigned)K, 128u, 32u, 128u, (unsigned)K);
      tdm_load_2d_bf16(lds_off(&lB[buf][0][0]),
                       gB + ((size_t)tileN * K + kb) * 2,
                       (unsigned)K, 128u, 32u, 128u, (unsigned)K);
    }
  };

  stage(0, 0);
  int buf = 0;
  for (int kb = 0; kb < K; kb += 32) {
    const bool has_next = (kb + 32) < K;
    if (has_next) stage(kb + 32, buf ^ 1);
    if (wave == 0) {
      if (has_next) __builtin_amdgcn_s_wait_tensorcnt(2);
      else          __builtin_amdgcn_s_wait_tensorcnt(0);
    }
    __syncthreads();

    v16bf af[2], bfr[4];
    for (int mi = 0; mi < 2; ++mi) {
      int m = waveM * 32 + mi * 16 + lanen;
      v8bf lo = *reinterpret_cast<const v8bf*>(&lA[buf][m][16 * half]);
      v8bf hi = *reinterpret_cast<const v8bf*>(&lA[buf][m][16 * half + 8]);
      af[mi] = make_frag(lo, hi);
    }
    for (int ni = 0; ni < 4; ++ni) {
      int n = waveN * 64 + ni * 16 + lanen;
      bfr[ni] = *reinterpret_cast<const v16bf*>(&lB[buf][n][16 * half]);
    }
    for (int mi = 0; mi < 2; ++mi)
      for (int ni = 0; ni < 4; ++ni)
        acc[mi][ni] = __builtin_amdgcn_wmma_f32_16x16x32_bf16(
            false, af[mi], false, bfr[ni], (short)0, acc[mi][ni], false, false);
    __syncthreads();
    buf ^= 1;
  }

  // epilogue: bias (+relu), store f32 or bf16
  for (int ni = 0; ni < 4; ++ni) {
    int col = tileN + waveN * 64 + ni * 16 + lanen;
    float bv = bias ? bias[col] : 0.f;
    for (int mi = 0; mi < 2; ++mi) {
      for (int r = 0; r < 8; ++r) {
        int row = tileM + waveM * 32 + mi * 16 + r + 8 * half;
        float v = acc[mi][ni][r] + bv;
        if (RELU) v = fmaxf(v, 0.f);
        if (OUT_BF16) Cb[(size_t)row * N + col] = (bf16_t)v;
        else          Cf[(size_t)row * N + col] = v;
      }
    }
  }
}

// ============================================================
// Flash-style attention, one wave per (b, h, 16-query tile).
// scores = (Q K^T)/8 + corr ; online softmax ; ctx = P V
// V chunk staged Global->LDS with async-to-LDS loads (ASYNCcnt).
// ============================================================
__global__ __launch_bounds__(32)
void attn_kernel(const bf16_t* __restrict__ Q, const bf16_t* __restrict__ Km,
                 const bf16_t* __restrict__ V, const float* __restrict__ corr,
                 bf16_t* __restrict__ O) {
  __shared__ __align__(64) bf16_t Pl[16][32];   // P tile (q x key-chunk)
  __shared__ __align__(64) bf16_t Vl[32][64];   // V chunk (key x d)
  const int lane  = threadIdx.x & 31;
  const int lanen = lane & 15;
  const int half  = lane >> 4;
  const int qtile = blockIdx.x;
  const int h     = blockIdx.y;
  const int b     = blockIdx.z;
  const int qbase = qtile * 16;
  const size_t rowbase = (size_t)b * SEQ;

  // Q fragments for d 0..31 and 32..63 (A layout), loaded once
  v16bf qf[2];
  {
    const bf16_t* qrow = Q + (rowbase + qbase + lanen) * D_MODEL + h * D_K;
    for (int ks = 0; ks < 2; ++ks) {
      v8bf lo = *reinterpret_cast<const v8bf*>(qrow + ks * 32 + 16 * half);
      v8bf hi = *reinterpret_cast<const v8bf*>(qrow + ks * 32 + 16 * half + 8);
      qf[ks] = make_frag(lo, hi);
    }
  }

  const v8f vzero = {0.f,0.f,0.f,0.f,0.f,0.f,0.f,0.f};
  v8f acc[4];
  for (int ni = 0; ni < 4; ++ni) acc[ni] = vzero;
  float rowmax[8], rowsum[8];
  for (int r = 0; r < 8; ++r) { rowmax[r] = -INFINITY; rowsum[r] = 0.f; }

  const unsigned vl_dst = lds_off(&Vl[lane][0]);

  for (int keybase = 0; keybase < SEQ; keybase += 32) {
    // ---- kick off async V-chunk stage (16B x 8 per lane, one key row each)
    {
      unsigned long long gsrc = (unsigned long long)(size_t)(
          V + (rowbase + keybase + lane) * D_MODEL + h * D_K);
      ASYNC_LDS_B128(vl_dst, gsrc, 0);
      ASYNC_LDS_B128(vl_dst, gsrc, 16);
      ASYNC_LDS_B128(vl_dst, gsrc, 32);
      ASYNC_LDS_B128(vl_dst, gsrc, 48);
      ASYNC_LDS_B128(vl_dst, gsrc, 64);
      ASYNC_LDS_B128(vl_dst, gsrc, 80);
      ASYNC_LDS_B128(vl_dst, gsrc, 96);
      ASYNC_LDS_B128(vl_dst, gsrc, 112);
    }
    // ---- scores: two 16x16 tiles over this 32-key chunk
    v8f sc[2] = {vzero, vzero};
    for (int t = 0; t < 2; ++t) {
      const bf16_t* krow = Km + (rowbase + keybase + t * 16 + lanen) * D_MODEL + h * D_K;
      for (int ks = 0; ks < 2; ++ks) {
        v16bf kf = *reinterpret_cast<const v16bf*>(krow + ks * 32 + 16 * half);
        sc[t] = __builtin_amdgcn_wmma_f32_16x16x32_bf16(
            false, qf[ks], false, kf, (short)0, sc[t], false, false);
      }
    }
    // ---- scale + additive correlation bias (+ prefetch next chunk's corr)
    float s0[8], s1[8];
    const float* cp = corr + ((size_t)b * SEQ + qbase) * SEQ + keybase;
    for (int r = 0; r < 8; ++r) {
      int rr = r + 8 * half;
      s0[r] = sc[0][r] * 0.125f + cp[(size_t)rr * SEQ + lanen];
      s1[r] = sc[1][r] * 0.125f + cp[(size_t)rr * SEQ + 16 + lanen];
    }
    if (keybase + 32 < SEQ) {
      for (int r = 0; r < 8; ++r)
        __builtin_prefetch(cp + (size_t)(r + 8 * half) * SEQ + 32 + lanen, 0, 1);
    }
    // ---- online softmax (row = r + 8*half lives in a 16-lane group)
    for (int r = 0; r < 8; ++r) {
      float m = fmaxf(s0[r], s1[r]);
      for (int mask = 1; mask < 16; mask <<= 1) m = fmaxf(m, __shfl_xor(m, mask, 32));
      float nm    = fmaxf(rowmax[r], m);
      float scale = __expf(rowmax[r] - nm);
      rowmax[r]   = nm;
      float p0 = __expf(s0[r] - nm);
      float p1 = __expf(s1[r] - nm);
      s0[r] = p0; s1[r] = p1;
      float ps = p0 + p1;
      for (int mask = 1; mask < 16; mask <<= 1) ps += __shfl_xor(ps, mask, 32);
      rowsum[r] = rowsum[r] * scale + ps;
      for (int ni = 0; ni < 4; ++ni) acc[ni][r] = acc[ni][r] * scale;
    }
    // ---- stage P (bf16) to LDS; join async V stage
    for (int r = 0; r < 8; ++r) {
      Pl[r + 8 * half][lanen]      = (bf16_t)s0[r];
      Pl[r + 8 * half][16 + lanen] = (bf16_t)s1[r];
    }
    ASYNC_WAIT0();
    __syncthreads();
    // ---- P as A-fragment, V columns as B-fragments, accumulate ctx
    v16bf pf;
    {
      v8bf lo = *reinterpret_cast<const v8bf*>(&Pl[lanen][16 * half]);
      v8bf hi = *reinterpret_cast<const v8bf*>(&Pl[lanen][16 * half + 8]);
      pf = make_frag(lo, hi);
    }
    for (int ni = 0; ni < 4; ++ni) {
      v16bf vf;
      for (int e = 0; e < 16; ++e) vf[e] = Vl[16 * half + e][ni * 16 + lanen];
      acc[ni] = __builtin_amdgcn_wmma_f32_16x16x32_bf16(
          false, pf, false, vf, (short)0, acc[ni], false, false);
    }
    __syncthreads();
  }

  // ---- normalize and store bf16 context (heads re-interleaved in place)
  for (int ni = 0; ni < 4; ++ni)
    for (int r = 0; r < 8; ++r) {
      int row = qbase + r + 8 * half;
      int col = h * D_K + ni * 16 + lanen;
      O[(rowbase + row) * D_MODEL + col] = (bf16_t)(acc[ni][r] / rowsum[r]);
    }
}

// ============================================================
// LayerNorm over rows of 1024: y = (x - mu) * rsqrt(var + eps) * g + be,
// x = base + delta. Optional f32 and bf16 outputs.
// ============================================================
__global__ __launch_bounds__(256)
void ln_kernel(const float* __restrict__ base, const float* __restrict__ delta,
               const float* __restrict__ g, const float* __restrict__ be,
               float* __restrict__ y32, bf16_t* __restrict__ y16) {
  __shared__ float s1[256];
  __shared__ float s2[256];
  const int row = blockIdx.x;
  const int tid = threadIdx.x;
  const float* pb = base  + (size_t)row * D_MODEL;
  const float* pd = delta + (size_t)row * D_MODEL;
  float x[4]; float sum = 0.f, sq = 0.f;
  for (int j = 0; j < 4; ++j) {
    int c = tid + j * 256;
    x[j] = pb[c] + pd[c];
    sum += x[j]; sq += x[j] * x[j];
  }
  s1[tid] = sum; s2[tid] = sq;
  __syncthreads();
  for (int off = 128; off > 0; off >>= 1) {
    if (tid < off) { s1[tid] += s1[tid + off]; s2[tid] += s2[tid + off]; }
    __syncthreads();
  }
  float mean = s1[0] * (1.f / D_MODEL);
  float var  = s2[0] * (1.f / D_MODEL) - mean * mean;
  float inv  = rsqrtf(var + 1e-5f);
  for (int j = 0; j < 4; ++j) {
    int c = tid + j * 256;
    float y = (x[j] - mean) * inv * g[c] + be[c];
    if (y32) y32[(size_t)row * D_MODEL + c] = y;
    if (y16) y16[(size_t)row * D_MODEL + c] = (bf16_t)y;
  }
}

// ============================================================
// Orchestration
// ============================================================
extern "C" void kernel_launch(void* const* d_in, const int* in_sizes, int n_in,
                              void* d_out, int out_size, void* d_ws, size_t ws_size,
                              hipStream_t stream) {
  (void)in_sizes; (void)n_in; (void)out_size; (void)ws_size;
  const float* src  = (const float*)d_in[0];
  const float* corr = (const float*)d_in[1];
  const float* pos  = (const float*)d_in[2];
  const float* Wq = (const float*)d_in[3];  const float* bq  = (const float*)d_in[4];
  const float* Wk = (const float*)d_in[5];  const float* bk  = (const float*)d_in[6];
  const float* Wv = (const float*)d_in[7];  const float* bv  = (const float*)d_in[8];
  const float* Wo = (const float*)d_in[9];  const float* bo  = (const float*)d_in[10];
  const float* W1 = (const float*)d_in[11]; const float* b1  = (const float*)d_in[12];
  const float* W2 = (const float*)d_in[13]; const float* b2  = (const float*)d_in[14];
  const float* g1 = (const float*)d_in[15]; const float* be1 = (const float*)d_in[16];
  const float* g2 = (const float*)d_in[17]; const float* be2 = (const float*)d_in[18];
  float* out = (float*)d_out;

  char* w = (char*)d_ws;
  const size_t MB = 1024 * 1024;
  bf16_t* qkin  = (bf16_t*)(w +   0 * MB);  // [4096,1024] bf16  8MB
  bf16_t* srcb  = (bf16_t*)(w +   8 * MB);  // [4096,1024] bf16  8MB
  bf16_t* WqT   = (bf16_t*)(w +  16 * MB);  // [1024,1024] bf16  2MB
  bf16_t* WkT   = (bf16_t*)(w +  18 * MB);
  bf16_t* WvT   = (bf16_t*)(w +  20 * MB);
  bf16_t* WoT   = (bf16_t*)(w +  22 * MB);
  bf16_t* W1T   = (bf16_t*)(w +  24 * MB);  // [4096,1024] bf16  8MB
  bf16_t* W2T   = (bf16_t*)(w +  32 * MB);  // [1024,4096] bf16  8MB
  bf16_t* Qb    = (bf16_t*)(w +  40 * MB);  // [4096,1024] bf16  8MB
  bf16_t* Kb    = (bf16_t*)(w +  48 * MB);
  bf16_t* Vb    = (bf16_t*)(w +  56 * MB);
  bf16_t* attnb = (bf16_t*)(w +  64 * MB);  // concat ctx bf16   8MB
  float*  aproj = (float*) (w +  72 * MB);  // attn proj f32    16MB
  float*  x1f   = (float*) (w +  88 * MB);  // LN1 out f32      16MB
  bf16_t* x1b   = (bf16_t*)(w + 104 * MB);  // LN1 out bf16      8MB
  bf16_t* ffh   = (bf16_t*)(w + 112 * MB);  // relu(x1 W1) bf16 32MB
  float*  ffn   = (float*) (w + 144 * MB);  // FFN out f32      16MB
  // total scratch: 160MB

  dim3 b256(256);
  // 1) activation prep
  prep_kernel<<<dim3((NROWS * D_MODEL) / 256), b256, 0, stream>>>(src, pos, qkin, srcb);
  // 2) weight transpose+convert
  transpose_kernel<<<dim3((D_MODEL * D_MODEL) / 256), b256, 0, stream>>>(Wq, WqT, D_MODEL, D_MODEL);
  transpose_kernel<<<dim3((D_MODEL * D_MODEL) / 256), b256, 0, stream>>>(Wk, WkT, D_MODEL, D_MODEL);
  transpose_kernel<<<dim3((D_MODEL * D_MODEL) / 256), b256, 0, stream>>>(Wv, WvT, D_MODEL, D_MODEL);
  transpose_kernel<<<dim3((D_MODEL * D_MODEL) / 256), b256, 0, stream>>>(Wo, WoT, D_MODEL, D_MODEL);
  transpose_kernel<<<dim3((D_MODEL * D_FF)    / 256), b256, 0, stream>>>(W1, W1T, D_MODEL, D_FF);
  transpose_kernel<<<dim3((D_FF * D_MODEL)    / 256), b256, 0, stream>>>(W2, W2T, D_FF, D_MODEL);
  // 3) Q/K/V projections (bf16 out)
  gemm_bf16_kernel<true, false><<<dim3(D_MODEL / 128, NROWS / 128), b256, 0, stream>>>(
      qkin, WqT, bq, nullptr, Qb, NROWS, D_MODEL, D_MODEL);
  gemm_bf16_kernel<true, false><<<dim3(D_MODEL / 128, NROWS / 128), b256, 0, stream>>>(
      qkin, WkT, bk, nullptr, Kb, NROWS, D_MODEL, D_MODEL);
  gemm_bf16_kernel<true, false><<<dim3(D_MODEL / 128, NROWS / 128), b256, 0, stream>>>(
      srcb, WvT, bv, nullptr, Vb, NROWS, D_MODEL, D_MODEL);
  // 4) attention
  attn_kernel<<<dim3(SEQ / 16, HEADS, BATCH), dim3(32), 0, stream>>>(Qb, Kb, Vb, corr, attnb);
  // 5) output projection (f32 out)
  gemm_bf16_kernel<false, false><<<dim3(D_MODEL / 128, NROWS / 128), b256, 0, stream>>>(
      attnb, WoT, bo, aproj, nullptr, NROWS, D_MODEL, D_MODEL);
  // 6) LN1 (residual with src)
  ln_kernel<<<dim3(NROWS), b256, 0, stream>>>(src, aproj, g1, be1, x1f, x1b);
  // 7) FFN
  gemm_bf16_kernel<true, true><<<dim3(D_FF / 128, NROWS / 128), b256, 0, stream>>>(
      x1b, W1T, b1, nullptr, ffh, NROWS, D_FF, D_MODEL);
  gemm_bf16_kernel<false, false><<<dim3(D_MODEL / 128, NROWS / 128), b256, 0, stream>>>(
      ffh, W2T, b2, ffn, nullptr, NROWS, D_MODEL, D_FF);
  // 8) LN2 -> final output
  ln_kernel<<<dim3(NROWS), b256, 0, stream>>>(x1f, ffn, g2, be2, out, nullptr);
}
